// GAN_3547642986904
// MI455X (gfx1250) — matrix operand order repaired and checked by reading, not compiled, not run on hardware
//
#include <hip/hip_runtime.h>
#include <stdint.h>

#define NN        8192
#define F_IN      256
#define F_OUT     64
#define NEG_SLOPE 0.01f

#define M_TILE    32      // output rows per block
#define K_CHUNK   128     // j-columns processed per iteration (4 WMMA K=32 steps)

typedef __attribute__((ext_vector_type(16))) __bf16 v16bf;
typedef __attribute__((ext_vector_type(8)))  float  v8f;

__device__ __forceinline__ uint32_t f2bits(float f) {
    union { float f; uint32_t u; } x; x.f = f; return x.u;
}
// round-to-nearest-even f32 -> bf16
__device__ __forceinline__ uint16_t f2bf16(float f) {
    uint32_t u = f2bits(f);
    u += 0x7FFFu + ((u >> 16) & 1u);
    return (uint16_t)(u >> 16);
}

// ---------------------------------------------------------------------------
// Kernel 1: Wh = H@W + bW ; s_src = Wh@a_w[:64] ; s_tar = Wh@a_w[64:]
// Writes Wh transposed in bf16 (WhT[F_OUT][NN]) so the main kernel's B-matrix
// fragments (K-major pairs) are contiguous.
// grid = NN blocks x 64 threads (one row per block, one output feature/thread)
// ---------------------------------------------------------------------------
__global__ __launch_bounds__(64) void gat_precompute(
    const float* __restrict__ H, const float* __restrict__ W,
    const float* __restrict__ bW, const float* __restrict__ a_w,
    uint16_t* __restrict__ WhT, float* __restrict__ s_src,
    float* __restrict__ s_tar)
{
    __shared__ __align__(16) float hrow[F_IN];
    __shared__ float red_s[F_OUT];
    __shared__ float red_t[F_OUT];
    const int i = blockIdx.x;
    const int t = threadIdx.x;                 // 0..63

    *(float4*)(hrow + t * 4) = *(const float4*)(H + (size_t)i * F_IN + t * 4);
    __syncthreads();

    float wh = bW[t];
#pragma unroll 8
    for (int k = 0; k < F_IN; ++k)
        wh = fmaf(hrow[k], W[k * F_OUT + t], wh);

    WhT[(size_t)t * NN + i] = f2bf16(wh);
    red_s[t] = wh * a_w[t];
    red_t[t] = wh * a_w[F_OUT + t];
    __syncthreads();
#pragma unroll
    for (int s = 32; s > 0; s >>= 1) {
        if (t < s) { red_s[t] += red_s[t + s]; red_t[t] += red_t[t + s]; }
        __syncthreads();
    }
    if (t == 0) { s_src[i] = red_s[0]; s_tar[i] = red_t[0]; }
}

// ---------------------------------------------------------------------------
// Kernel 2: fused  e-generation + denom (e*A row sums) + (e @ Wh) via WMMA
//           + sigmoid.  One block = 32 output rows x all 64 columns.
// 256 threads = 8 waves: wave = (mw in 0..1)*4 + (nw in 0..3)
//   wave computes C tile rows [mw*16,mw*16+16) x cols [nw*16,nw*16+16)
// grid = NN/M_TILE = 256 blocks
// ---------------------------------------------------------------------------
__global__ __launch_bounds__(256) void gat_fused(
    const int* __restrict__ A,
    const float* __restrict__ s_src, const float* __restrict__ s_tar,
    const float* __restrict__ a_b_p,
    const uint16_t* __restrict__ WhT,
    float* __restrict__ out)
{
    __shared__ __align__(16) uint16_t e_sh[M_TILE * K_CHUNK];   // 8 KB  (bf16 e-tile, row-major)
    __shared__ __align__(16) uint16_t bt_sh[F_OUT * K_CHUNK];   // 16 KB (WhT tile: [n][k])
    __shared__ float red[256];
    __shared__ float denom_sh[M_TILE];

    const int tid  = threadIdx.x;
    const int lane = tid & 31;
    const int wave = tid >> 5;           // 0..7
    const int mw   = wave >> 2;          // 0..1 : which 16-row half
    const int nw   = wave & 3;           // 0..3 : which 16-col slice
    const int i0   = blockIdx.x * M_TILE;

    const float a_b = a_b_p[0];

    // phase-1 mapping: e-tile + A streaming. Thread owns row r, 16 cols at c0.
    const int r  = tid >> 3;             // 0..31
    const int c0 = (tid & 7) * 16;       // 0,16,...,112
    const float ssrc_r = s_src[i0 + r] + a_b;
    const int* Arow = A + (size_t)(i0 + r) * NN;

    // WhT staging mapping: thread copies 32 bf16 of row bn
    const int bn = tid >> 2;             // 0..63
    const int bh = (tid & 3) * 32;       // 0,32,64,96
    const uint16_t* wtrow = WhT + (size_t)bn * NN;

    // WMMA fragment lane coordinates (16-bit layouts, wave32)
    const int M  = lane & 15;
    const int hi = (lane >> 4) & 1;      // lane half selects K half
    const uint16_t* arow = e_sh  + (mw * 16 + M) * K_CHUNK;  // A frag: row M
    const uint16_t* brow = bt_sh + (nw * 16 + M) * K_CHUNK;  // B frag: col N=M

    float dpart = 0.f;
    v8f acc = {};

    for (int j = 0; j < NN; j += K_CHUNK) {
        // ---- stage WhT[n][j..j+128) -> LDS (coalesced b128 loads) ----
#pragma unroll
        for (int q = 0; q < 4; ++q) {
            *(uint4*)(bt_sh + bn * K_CHUNK + bh + q * 8) =
                *(const uint4*)(wtrow + j + bh + q * 8);
        }
        // ---- generate e tile (once), accumulate denom, pack bf16 ----
#pragma unroll
        for (int g = 0; g < 4; ++g) {
            const int cc = c0 + g * 4;
            float4 st = *(const float4*)(s_tar + j + cc);
            int4   ad = *(const int4*)(Arow + j + cc);
            float x0 = ssrc_r + st.x; x0 = x0 >= 0.f ? x0 : NEG_SLOPE * x0;
            float x1 = ssrc_r + st.y; x1 = x1 >= 0.f ? x1 : NEG_SLOPE * x1;
            float x2 = ssrc_r + st.z; x2 = x2 >= 0.f ? x2 : NEG_SLOPE * x2;
            float x3 = ssrc_r + st.w; x3 = x3 >= 0.f ? x3 : NEG_SLOPE * x3;
            float e0 = __expf(x0), e1 = __expf(x1), e2 = __expf(x2), e3 = __expf(x3);
            dpart += e0 * (float)ad.x + e1 * (float)ad.y +
                     e2 * (float)ad.z + e3 * (float)ad.w;
            uint32_t p01 = (uint32_t)f2bf16(e0) | ((uint32_t)f2bf16(e1) << 16);
            uint32_t p23 = (uint32_t)f2bf16(e2) | ((uint32_t)f2bf16(e3) << 16);
            *(uint32_t*)(e_sh + r * K_CHUNK + cc)     = p01;
            *(uint32_t*)(e_sh + r * K_CHUNK + cc + 2) = p23;
        }
        // prefetch next A chunk (256 threads x 64B line == next 16 KB tile)
        if (j + K_CHUNK < NN)
            __builtin_prefetch(Arow + j + K_CHUNK + c0, 0, 0);
        __syncthreads();

        // ---- 4 x v_wmma_f32_16x16x32_bf16 per wave ----
#pragma unroll
        for (int kk = 0; kk < 4; ++kk) {
            union { uint32_t u[8]; v16bf v; } afrag, bfrag;
#pragma unroll
            for (int v = 0; v < 8; ++v) {
                // A 16x32 bf16 layout: v<4 -> K = 2v(+8 for hi lanes); v>=4 -> 16+2(v-4)(+8)
                const int ka = ((v < 4) ? (2 * v) : (8 + 2 * v)) + hi * 8;
                afrag.u[v] = *(const uint32_t*)(arow + kk * 32 + ka);
                // B 32x16 bf16 layout: same K order as A: v<4 -> K = 2v(+8 for hi lanes); v>=4 -> 16+2(v-4)(+8)
                const int kb = ((v < 4) ? (2 * v) : (8 + 2 * v)) + hi * 8;
                bfrag.u[v] = *(const uint32_t*)(brow + kk * 32 + kb);
            }
            acc = __builtin_amdgcn_wmma_f32_16x16x32_bf16(
                false, afrag.v, false, bfrag.v, (short)0, acc, false, false);
        }
        __syncthreads();
    }

    // ---- reduce denominator partials: 8 threads per row ----
    red[tid] = dpart;
    __syncthreads();
    if (tid < M_TILE) {
        float s = 0.f;
#pragma unroll
        for (int q = 0; q < 8; ++q) s += red[tid * 8 + q];
        denom_sh[tid] = s;
    }
    __syncthreads();

    // ---- out = sigmoid(acc / denom)  (C/D layout: VGPR v -> M = v + hi*8) ----
#pragma unroll
    for (int v = 0; v < 8; ++v) {
        const int m = mw * 16 + v + hi * 8;
        const float val = acc[v] / denom_sh[m];
        out[(size_t)(i0 + m) * F_OUT + nw * 16 + M] = 1.f / (1.f + __expf(-val));
    }
}

// ---------------------------------------------------------------------------
extern "C" void kernel_launch(void* const* d_in, const int* in_sizes, int n_in,
                              void* d_out, int out_size, void* d_ws, size_t ws_size,
                              hipStream_t stream) {
    (void)in_sizes; (void)n_in; (void)out_size; (void)ws_size;
    const float* H   = (const float*)d_in[0];
    const int*   A   = (const int*)d_in[1];
    const float* W   = (const float*)d_in[2];
    const float* bW  = (const float*)d_in[3];
    const float* a_w = (const float*)d_in[4];
    const float* a_b = (const float*)d_in[5];
    float* out = (float*)d_out;

    uint8_t* ws = (uint8_t*)d_ws;
    uint16_t* WhT  = (uint16_t*)ws;                                  // 1 MB
    float*    ssrc = (float*)(ws + (size_t)F_OUT * NN * sizeof(uint16_t));
    float*    star = ssrc + NN;                                      // +32 KB each

    gat_precompute<<<NN, 64, 0, stream>>>(H, W, bW, a_w, WhT, ssrc, star);
    gat_fused<<<NN / M_TILE, 256, 0, stream>>>(A, ssrc, star, a_b, WhT, out);
}
